// LSTMModel_41790031790159
// MI455X (gfx1250) — matrix operand order repaired and checked by reading, not compiled, not run on hardware
//
#include <hip/hip_runtime.h>
#include <hip/hip_bf16.h>

// ---------------------------------------------------------------------------
// LSTM (T=8192, I=256, H=512, O=10000) for MI455X / gfx1250 (wave32, WMMA).
//   1) cvt fp32->bf16: X, W_ih, W_lin
//   2) x_gates = X @ W_ih^T + (b_ih+b_hh)   bf16 WMMA, async-to-LDS pipeline
//   3) persistent 8-WG scan, W_hh bf16 in 256KB LDS slices, atomic grid barrier
//   4) logits = hs @ W_lin^T + b_lin        bf16 WMMA, async-to-LDS pipeline
//   5) row-wise log_softmax in place on d_out
// ---------------------------------------------------------------------------

#define T_STEPS 8192
#define IN_DIM  256
#define HID     512
#define GATES   2048
#define OUT_DIM 10000

typedef __bf16 v16bf __attribute__((ext_vector_type(16)));
typedef float  v8f   __attribute__((ext_vector_type(8)));

struct __align__(16) B128 { unsigned int x, y, z, w; };

static __device__ __forceinline__ unsigned short f32_to_bf16_bits(float f) {
    unsigned u = __float_as_uint(f);
    u += 0x7FFFu + ((u >> 16) & 1u);   // round-to-nearest-even
    return (unsigned short)(u >> 16);
}

// ------------------------------- conversion --------------------------------
__global__ void cvt_f32_bf16_kernel(const float* __restrict__ in,
                                    unsigned short* __restrict__ out, int n) {
    for (int i = blockIdx.x * blockDim.x + threadIdx.x; i < n;
         i += gridDim.x * blockDim.x)
        out[i] = f32_to_bf16_bits(in[i]);
}

__global__ void init_barrier_kernel(unsigned* bar) {
    if (threadIdx.x == 0 && blockIdx.x == 0) *bar = 0u;
}

// ------------------------------- WMMA GEMM ---------------------------------
// C[M,N] = A[M,K] * B[N,K]^T (+bias0[n] +bias1[n]); A/B bf16 row-major, C f32.
// 256 threads = 8 waves (4 M x 2 N); wave tile 32x64 -> 8 WMMA / K-step.
// Tiles staged with GLOBAL_LOAD_ASYNC_TO_LDS_B128, double buffered (ASYNCcnt).
#define BM 128
#define BN 128
#define BK 32

// A fragment (16x32 bf16): lane<16 -> K {0..7,16..23}, lane>=16 -> K {8..15,24..31}
static __device__ __forceinline__ v16bf load_fragA(const unsigned short* tile,
                                                   int row0, int lane) {
    int r  = row0 + (lane & 15);
    int kh = (lane & 16) ? 8 : 0;
    union { v16bf v; B128 q[2]; } f;
    f.q[0] = *(const B128*)(tile + r * BK + kh);
    f.q[1] = *(const B128*)(tile + r * BK + kh + 16);
    return f.v;
}
// B fragment (32x16 bf16): lane<16 -> K {0..7,16..23} of column n, lane>=16 -> K {8..15,24..31}
static __device__ __forceinline__ v16bf load_fragB(const unsigned short* tile,
                                                   int col0, int lane) {
    int n  = col0 + (lane & 15);
    int kh = (lane & 16) ? 8 : 0;
    union { v16bf v; B128 q[2]; } f;
    f.q[0] = *(const B128*)(tile + n * BK + kh);
    f.q[1] = *(const B128*)(tile + n * BK + kh + 16);
    return f.v;
}

static __device__ __forceinline__ v8f wmma_bf16(v16bf a, v16bf b, v8f c) {
    return __builtin_amdgcn_wmma_f32_16x16x32_bf16(
        false, a, false, b, (short)0, c, false, false);
}

// one 16B async global->LDS copy (both addresses share INST_OFFSET semantics)
static __device__ __forceinline__ void async_b128(unsigned lds_off,
                                                  unsigned long long gaddr) {
    asm volatile("global_load_async_to_lds_b128 %0, %1, off"
                 :: "v"(lds_off), "v"(gaddr) : "memory");
}
static __device__ __forceinline__ void wait_async0() {
    asm volatile("s_wait_asynccnt 0" ::: "memory");
}

__global__ __launch_bounds__(256)
void gemm_bf16_wmma_kernel(const unsigned short* __restrict__ A,
                           const unsigned short* __restrict__ B,
                           float* __restrict__ C,
                           const float* __restrict__ bias0,
                           const float* __restrict__ bias1,
                           int M, int N, int K) {
    __shared__ __align__(16) unsigned short As[2][BM * BK];   // 2 x 8 KB
    __shared__ __align__(16) unsigned short Bs[2][BN * BK];   // 2 x 8 KB

    const int tid  = threadIdx.x;
    const int lane = tid & 31;
    const int wave = tid >> 5;
    const int wm   = wave >> 1;         // 0..3  (M)
    const int wn   = wave & 1;          // 0..1  (N)
    const int mBase = blockIdx.y * BM;
    const int nBase = blockIdx.x * BN;

    v8f zero = {};
    v8f acc[2][4];
#pragma unroll
    for (int im = 0; im < 2; ++im)
#pragma unroll
        for (int in_ = 0; in_ < 4; ++in_) acc[im][in_] = zero;

    // staging map: 2 threads per 32-half row, 32B (2 x b128) per thread
    const int ar = tid >> 1, aseg = tid & 1;
    const int bnrow = min(nBase + ar, N - 1);   // clamp: cols >= N never stored

    const unsigned aOff[2] = {
        (unsigned)(size_t)&As[0][ar * BK + aseg * 16],
        (unsigned)(size_t)&As[1][ar * BK + aseg * 16] };
    const unsigned bOff[2] = {
        (unsigned)(size_t)&Bs[0][ar * BK + aseg * 16],
        (unsigned)(size_t)&Bs[1][ar * BK + aseg * 16] };
    const unsigned short* aRow = A + (size_t)(mBase + ar) * K + aseg * 16;
    const unsigned short* bRow = B + (size_t)bnrow * K + aseg * 16;

    auto stage = [&](int k0, int buf) {
        unsigned long long ga = (unsigned long long)(size_t)(aRow + k0);
        unsigned long long gb = (unsigned long long)(size_t)(bRow + k0);
        asm volatile("global_load_async_to_lds_b128 %0, %2, off\n\t"
                     "global_load_async_to_lds_b128 %0, %2, off offset:16\n\t"
                     "global_load_async_to_lds_b128 %1, %3, off\n\t"
                     "global_load_async_to_lds_b128 %1, %3, off offset:16"
                     :: "v"(aOff[buf]), "v"(bOff[buf]), "v"(ga), "v"(gb)
                     : "memory");
    };

    const int nk = K / BK;
    stage(0, 0);
    wait_async0();
    __syncthreads();

    for (int i = 0; i < nk; ++i) {
        const int cur = i & 1;
        if (i + 1 < nk) stage((i + 1) * BK, cur ^ 1);

        v16bf a0 = load_fragA(As[cur], wm * 32 + 0,  lane);
        v16bf a1 = load_fragA(As[cur], wm * 32 + 16, lane);
#pragma unroll
        for (int in_ = 0; in_ < 4; ++in_) {
            v16bf b = load_fragB(Bs[cur], wn * 64 + in_ * 16, lane);
            acc[0][in_] = wmma_bf16(a0, b, acc[0][in_]);
            acc[1][in_] = wmma_bf16(a1, b, acc[1][in_]);
        }
        if (i + 1 < nk) wait_async0();
        __syncthreads();
    }

    // epilogue: D layout lane<16 -> {N=lane, M=r}, lane>=16 -> {N=lane-16, M=8+r}
    for (int im = 0; im < 2; ++im)
        for (int in_ = 0; in_ < 4; ++in_) {
            int n = nBase + wn * 64 + in_ * 16 + (lane & 15);
            if (n >= N) continue;
            float bv = 0.f;
            if (bias0) bv += bias0[n];
            if (bias1) bv += bias1[n];
            int m0 = mBase + wm * 32 + im * 16 + ((lane < 16) ? 0 : 8);
#pragma unroll
            for (int r = 0; r < 8; ++r)
                C[(size_t)(m0 + r) * N + n] = acc[im][in_][r] + bv;
        }
}

// ------------------------------- LSTM scan ---------------------------------
#define SCAN_WGS 8
#define SCAN_THR 512
#define UNITS_PER_WG 64
#define ROWS_PER_WG 256
#define SCAN_SMEM_BYTES (ROWS_PER_WG * HID * 2 + HID * 4 + ROWS_PER_WG * 4)

__global__ __launch_bounds__(SCAN_THR)
void lstm_scan_kernel(const float* __restrict__ xg,       // [T, 4H] fp32
                      const float* __restrict__ Whh,      // [4H, H] fp32
                      unsigned short* __restrict__ hs_b,  // [T, H] bf16 out
                      float* __restrict__ hg,             // [H] fp32 h exchange
                      unsigned* __restrict__ bar) {
    extern __shared__ char smem[];
    unsigned short* Ws = (unsigned short*)smem;                          // [256][512]
    float* h_s    = (float*)(smem + ROWS_PER_WG * HID * 2);              // [512]
    float* gate_s = (float*)(smem + ROWS_PER_WG * HID * 2 + HID * 4);    // [256]

    const int tid = threadIdx.x;
    const int wg  = blockIdx.x;

    for (int idx = tid; idx < ROWS_PER_WG * HID; idx += SCAN_THR) {
        int r = idx >> 9;
        int k = idx & (HID - 1);
        int p = r >> 6, u = r & 63;
        int grow = p * HID + wg * UNITS_PER_WG + u;
        Ws[idx] = f32_to_bf16_bits(Whh[(size_t)grow * HID + k]);
    }
    float c = 0.f;
    __syncthreads();

    const int r    = tid >> 1;
    const int half = tid & 1;

    for (int t = 0; t < T_STEPS; ++t) {
        h_s[tid] = (t == 0) ? 0.f : hg[tid];
        __syncthreads();

        const unsigned short* wrow = Ws + r * HID + half * 256;
        const float* hp = h_s + half * 256;
        float sum = 0.f;
#pragma unroll 8
        for (int k = 0; k < 256; k += 2) {
            unsigned pk = *(const unsigned*)(wrow + k);
            float w0 = __uint_as_float((pk & 0xFFFFu) << 16);
            float w1 = __uint_as_float(pk & 0xFFFF0000u);
            float2 hv = *(const float2*)(hp + k);
            sum = fmaf(w0, hv.x, sum);
            sum = fmaf(w1, hv.y, sum);
        }
        sum += __shfl_xor(sum, 1, 32);
        if (half == 0) {
            int p = r >> 6, u = r & 63;
            int grow = p * HID + wg * UNITS_PER_WG + u;
            gate_s[r] = sum + xg[(size_t)t * GATES + grow];
        }
        __syncthreads();

        if (tid < UNITS_PER_WG) {
            float gi = gate_s[tid];
            float gf = gate_s[UNITS_PER_WG + tid];
            float gg = gate_s[2 * UNITS_PER_WG + tid];
            float go = gate_s[3 * UNITS_PER_WG + tid];
            float i_ = 1.f / (1.f + __expf(-gi));
            float f_ = 1.f / (1.f + __expf(-gf));
            float g_ = tanhf(gg);
            float o_ = 1.f / (1.f + __expf(-go));
            c = f_ * c + i_ * g_;
            float h = o_ * tanhf(c);
            int j = wg * UNITS_PER_WG + tid;
            hg[j] = h;
            hs_b[(size_t)t * HID + j] = f32_to_bf16_bits(h);
        }
        __syncthreads();

        if (tid == 0) {
            __threadfence();
            __hip_atomic_fetch_add(bar, 1u, __ATOMIC_ACQ_REL, __HIP_MEMORY_SCOPE_AGENT);
            unsigned target = (unsigned)(t + 1) * SCAN_WGS;
            while (__hip_atomic_load(bar, __ATOMIC_ACQUIRE, __HIP_MEMORY_SCOPE_AGENT) < target)
                __builtin_amdgcn_s_sleep(1);
        }
        __syncthreads();
    }
}

// ------------------------------ log_softmax --------------------------------
__global__ __launch_bounds__(256)
void logsoftmax_kernel(float* __restrict__ out, int N) {
    __shared__ float red[256];
    const int tid = threadIdx.x;
    float* p = out + (size_t)blockIdx.x * N;

    float m = -INFINITY;
    for (int i = tid; i < N; i += 256) m = fmaxf(m, p[i]);
    red[tid] = m; __syncthreads();
    for (int s = 128; s > 0; s >>= 1) {
        if (tid < s) red[tid] = fmaxf(red[tid], red[tid + s]);
        __syncthreads();
    }
    m = red[0]; __syncthreads();

    float sum = 0.f;
    for (int i = tid; i < N; i += 256) sum += expf(p[i] - m);
    red[tid] = sum; __syncthreads();
    for (int s = 128; s > 0; s >>= 1) {
        if (tid < s) red[tid] += red[tid + s];
        __syncthreads();
    }
    float ls = m + logf(red[0]);
    for (int i = tid; i < N; i += 256) p[i] = p[i] - ls;
}

// ------------------------------ launch glue --------------------------------
#define OFF_XG    ((size_t)0)
#define OFF_XB    (OFF_XG   + (size_t)T_STEPS * GATES * 4)
#define OFF_WIHB  (OFF_XB   + (size_t)T_STEPS * IN_DIM * 2)
#define OFF_WLINB (OFF_WIHB + (size_t)GATES * IN_DIM * 2)
#define OFF_HSB   (OFF_WLINB + (size_t)OUT_DIM * HID * 2)
#define OFF_HG    (OFF_HSB  + (size_t)T_STEPS * HID * 2)
#define OFF_BAR   (OFF_HG   + (size_t)HID * 4)

extern "C" void kernel_launch(void* const* d_in, const int* in_sizes, int n_in,
                              void* d_out, int out_size, void* d_ws, size_t ws_size,
                              hipStream_t stream) {
    const float* X     = (const float*)d_in[0];
    const float* W_ih  = (const float*)d_in[1];
    const float* W_hh  = (const float*)d_in[2];
    const float* b_ih  = (const float*)d_in[3];
    const float* b_hh  = (const float*)d_in[4];
    const float* W_lin = (const float*)d_in[5];
    const float* b_lin = (const float*)d_in[6];
    float* out = (float*)d_out;

    char* ws = (char*)d_ws;
    float*          xg    = (float*)(ws + OFF_XG);
    unsigned short* Xb    = (unsigned short*)(ws + OFF_XB);
    unsigned short* Wihb  = (unsigned short*)(ws + OFF_WIHB);
    unsigned short* Wlinb = (unsigned short*)(ws + OFF_WLINB);
    unsigned short* hs_b  = (unsigned short*)(ws + OFF_HSB);
    float*          hg    = (float*)(ws + OFF_HG);
    unsigned*       bar   = (unsigned*)(ws + OFF_BAR);

    cvt_f32_bf16_kernel<<<1024, 256, 0, stream>>>(X, Xb, T_STEPS * IN_DIM);
    cvt_f32_bf16_kernel<<<1024, 256, 0, stream>>>(W_ih, Wihb, GATES * IN_DIM);
    cvt_f32_bf16_kernel<<<1024, 256, 0, stream>>>(W_lin, Wlinb, OUT_DIM * HID);
    init_barrier_kernel<<<1, 32, 0, stream>>>(bar);

    gemm_bf16_wmma_kernel<<<dim3(GATES / BN, T_STEPS / BM), 256, 0, stream>>>(
        Xb, Wihb, xg, b_ih, b_hh, T_STEPS, GATES, IN_DIM);

    lstm_scan_kernel<<<SCAN_WGS, SCAN_THR, SCAN_SMEM_BYTES, stream>>>(
        xg, W_hh, hs_b, hg, bar);

    gemm_bf16_wmma_kernel<<<dim3((OUT_DIM + BN - 1) / BN, T_STEPS / BM), 256, 0, stream>>>(
        hs_b, Wlinb, out, b_lin, nullptr, T_STEPS, OUT_DIM, HID);

    logsoftmax_kernel<<<T_STEPS, 256, 0, stream>>>(out, OUT_DIM);
}
